// CausalMultiheadAttention_19473381720564
// MI455X (gfx1250) — compile-verified, matrix operand order checked by reading, not run on hardware
//
#include <hip/hip_runtime.h>

#define BB 4
#define SS 1024
#define CC 3072
#define EE 512
#define HH 8
#define DD 64
#define KVL (CC + SS)   // 4096

typedef __attribute__((ext_vector_type(16))) __bf16 v16bf;
typedef __attribute__((ext_vector_type(8)))  float  v8f;

union Frag { v16bf v; unsigned int u[8]; };

__device__ __forceinline__ unsigned short f2bf(float f) {
  unsigned int u = __float_as_uint(f);
  u = (u + 0x7FFFu + ((u >> 16) & 1u)) >> 16;
  return (unsigned short)u;
}

// load one 16x32 bf16 A/B fragment (pairs along K contiguous at `base`)
__device__ __forceinline__ void load_frag(Frag& f, const unsigned short* base, int khalf) {
#pragma unroll
  for (int i = 0; i < 8; ++i) {
    int kb = ((i >> 2) << 4) + (khalf << 3) + ((i & 3) << 1);
    f.u[i] = *(const unsigned int*)(base + kb);
  }
}

// async DMA: 16 bytes/lane global -> LDS, tracked on ASYNCcnt
__device__ __forceinline__ void async_load_b128(unsigned lds_addr, const void* gptr) {
  asm volatile("global_load_async_to_lds_b128 %0, %1, off"
               :: "v"(lds_addr), "v"((unsigned long long)(size_t)gptr)
               : "memory");
}
__device__ __forceinline__ void wait_async0() {
#if __has_builtin(__builtin_amdgcn_s_wait_asynccnt)
  __builtin_amdgcn_s_wait_asynccnt(0);
#else
  asm volatile("s_wait_asynccnt 0x0" ::: "memory");
#endif
}
__device__ __forceinline__ void wait_async8() {
#if __has_builtin(__builtin_amdgcn_s_wait_asynccnt)
  __builtin_amdgcn_s_wait_asynccnt(8);
#else
  asm volatile("s_wait_asynccnt 0x8" ::: "memory");
#endif
}

// ---------------- f32 -> bf16 conversion kernels ----------------
__global__ void cvt_f32_bf16(const float* __restrict__ in,
                             unsigned short* __restrict__ out, long long n) {
  long long i = (long long)blockIdx.x * blockDim.x + threadIdx.x;
  long long stride = (long long)gridDim.x * blockDim.x;
  for (; i < n; i += stride) out[i] = f2bf(in[i]);
}

// (B,C,E) f32 -> first C rows of (B,KV,E) bf16
__global__ void cvt_cache(const float* __restrict__ in,
                          unsigned short* __restrict__ out, long long n) {
  long long i = (long long)blockIdx.x * blockDim.x + threadIdx.x;
  long long stride = (long long)gridDim.x * blockDim.x;
  const long long perB = (long long)CC * EE;
  for (; i < n; i += stride) {
    long long b = i / perB;
    long long r = i - b * perB;
    out[b * (long long)KVL * EE + r] = f2bf(in[i]);
  }
}

// ---------------- bf16 WMMA GEMM: Y = A * W^T (+bias) * scale ----------------
// A: MxK row-major bf16.  W: NxK row-major bf16 (B[k][n] = W[n*K+k]).
// Each wave: 16(m) x 64(n) tile, double-buffered fragment pipeline.
// mode 0: f32 contiguous MxN.  mode 1: bf16 contiguous MxN.
// mode 2: bf16 scattered: row m -> (b=m/S) row (C + m%S) of a (B,KV,N) tensor.
__global__ __launch_bounds__(256) void gemm_bf16(
    const unsigned short* __restrict__ A,
    const unsigned short* __restrict__ W,
    const float* __restrict__ bias,
    void* __restrict__ out,
    int M, int N, int K, int mode, float scale)
{
  int wave = (int)((blockIdx.x * blockDim.x + threadIdx.x) >> 5);
  int lane = threadIdx.x & 31;
  int ngroups = N >> 6;                      // 64-wide n groups
  int mt = wave / ngroups, ng = wave % ngroups;
  if (mt * 16 >= M) return;

  int rc    = lane & 15;   // row (for A) / col (for B and C/D)
  int khalf = lane >> 4;

  const unsigned short* arow  = A + (size_t)(mt * 16 + rc) * K;
  const unsigned short* brow0 = W + (size_t)(ng * 64 + rc) * K;

  Frag a0, a1, b0[4], b1[4];
  v8f acc[4] = {v8f{}, v8f{}, v8f{}, v8f{}};

  // preload k-step 0
  load_frag(a0, arow, khalf);
#pragma unroll
  for (int j = 0; j < 4; ++j) load_frag(b0[j], brow0 + (size_t)j * 16 * K, khalf);

  for (int kk = 0; kk < K; kk += 64) {
    bool has1 = (kk + 32 < K);
    if (has1) {                               // prefetch k-step kk+32 into buf1
      load_frag(a1, arow + kk + 32, khalf);
#pragma unroll
      for (int j = 0; j < 4; ++j)
        load_frag(b1[j], brow0 + (size_t)j * 16 * K + kk + 32, khalf);
    }
#pragma unroll
    for (int j = 0; j < 4; ++j)
      acc[j] = __builtin_amdgcn_wmma_f32_16x16x32_bf16(false, a0.v, false, b0[j].v,
                                                       (short)0, acc[j], false, false);
    if (has1) {
      if (kk + 64 < K) {                      // prefetch k-step kk+64 into buf0
        load_frag(a0, arow + kk + 64, khalf);
#pragma unroll
        for (int j = 0; j < 4; ++j)
          load_frag(b0[j], brow0 + (size_t)j * 16 * K + kk + 64, khalf);
      }
#pragma unroll
      for (int j = 0; j < 4; ++j)
        acc[j] = __builtin_amdgcn_wmma_f32_16x16x32_bf16(false, a1.v, false, b1[j].v,
                                                         (short)0, acc[j], false, false);
    }
  }

#pragma unroll
  for (int j = 0; j < 4; ++j) {
    int col = ng * 64 + j * 16 + rc;
    float bv = bias ? bias[col] : 0.f;
#pragma unroll
    for (int r = 0; r < 8; ++r) {
      int grow = mt * 16 + r + (khalf << 3);
      float val = (acc[j][r] + bv) * scale;
      if (mode == 0) {
        ((float*)out)[(size_t)grow * N + col] = val;
      } else if (mode == 1) {
        ((unsigned short*)out)[(size_t)grow * N + col] = f2bf(val);
      } else {
        int b = grow / SS, s = grow % SS;
        ((unsigned short*)out)[((size_t)b * KVL + CC + s) * (size_t)N + col] = f2bf(val);
      }
    }
  }
}

// ---------------- flash attention: one wave = 16 queries of one (b,h) ----------
// K fragments: per-lane b128 loads, double-buffered one kv-tile ahead.
// V tiles: async-DMA'd to a double-buffered LDS tile one iteration ahead;
// the PV stage only waits asynccnt<=8, hiding the DMA behind a full iteration.
__global__ __launch_bounds__(256) void attn_kernel(
    const unsigned short* __restrict__ Q,   // (B,S,E) bf16, pre-scaled by D^-0.5
    const unsigned short* __restrict__ Kc,  // (B,KV,E) bf16
    const unsigned short* __restrict__ Vc,  // (B,KV,E) bf16
    unsigned short* __restrict__ O)         // (B,S,E) bf16
{
  // per wave: P staging (512) + two 32x64 V tiles (2x2048) ushorts = 9KB
  __shared__ __align__(16) unsigned short lds[8 * 4608];
  int wslot = threadIdx.x >> 5;
  int wave  = (int)((blockIdx.x * blockDim.x + threadIdx.x) >> 5);
  int lane  = threadIdx.x & 31;

  const int stiles = SS / 16;                  // 64
  int b   = wave / (HH * stiles);
  int rem = wave % (HH * stiles);
  int h   = rem / stiles;
  int s0  = (rem % stiles) * 16;

  int rc    = lane & 15;
  int khalf = lane >> 4;

  unsigned short* plds  = lds + wslot * 4608;
  unsigned short* vlds0 = plds + 512;
  unsigned short* vlds1 = vlds0 + 2048;
  unsigned vbase0 = (unsigned)(size_t)(void*)vlds0;   // LDS byte offsets (flat low32)
  unsigned vbase1 = (unsigned)(size_t)(void*)vlds1;

  // preload Q A-fragments: D=64 -> two 16x32 fragments
  Frag qa[2];
  const unsigned short* qrow = Q + ((size_t)(b * SS + s0 + rc)) * EE + h * DD;
  load_frag(qa[0], qrow, khalf);
  load_frag(qa[1], qrow + 32, khalf);

  float mrow[8], lrow[8];
  v8f oacc[4] = {v8f{}, v8f{}, v8f{}, v8f{}};
#pragma unroll
  for (int r = 0; r < 8; ++r) { mrow[r] = -3.0e38f; lrow[r] = 0.f; }

  const unsigned short* kbase = Kc + (size_t)b * KVL * EE + h * DD;
  const unsigned short* vbase = Vc + (size_t)b * KVL * EE + h * DD;
  int kv_end = CC + s0 + 16;                   // exclusive causal bound

  // per-lane slice of the V tile DMA: lane -> (4 rows/instr, 8 lanes/row)
  int vrow_sub = lane >> 3;                    // 0..3
  int vcol     = (lane & 7) * 8;               // 8 ushorts = 16B per lane

  auto dmaV = [&](int kv0, unsigned ldsb) {
#pragma unroll
    for (int t = 0; t < 8; ++t) {
      int row = t * 4 + vrow_sub;
      async_load_b128(ldsb + (unsigned)(row * 64 + vcol) * 2u,
                      vbase + (size_t)(kv0 + row) * EE + vcol);
    }
  };
  auto loadK = [&](Frag (&kf)[4], int kv0) {
#pragma unroll
    for (int nt = 0; nt < 2; ++nt) {
      const unsigned short* krow = kbase + (size_t)(kv0 + nt * 16 + rc) * EE;
      load_frag(kf[nt * 2 + 0], krow, khalf);
      load_frag(kf[nt * 2 + 1], krow + 32, khalf);
    }
  };

  // one full kv-iteration: scores/softmax on kf, PV from V tile in LDS (vcur).
  // Issues the DMA for kv0+32 into vnext at the top when `more` is true.
  auto process = [&](int kv0, Frag (&kf)[4], unsigned short* vcur,
                     unsigned vnext, bool more) {
    if (more) dmaV(kv0 + 32, vnext);
    if (kv0 + 64 < kv_end)
      __builtin_prefetch(kbase + (size_t)(kv0 + 64 + rc) * EE, 0, 3);

    // ---- scores: 16 queries x 32 kv, K-dim = D = 64 ----
    v8f sacc[2];
#pragma unroll
    for (int nt = 0; nt < 2; ++nt) {
      v8f z = {};
      z = __builtin_amdgcn_wmma_f32_16x16x32_bf16(false, qa[0].v, false, kf[nt * 2 + 0].v,
                                                  (short)0, z, false, false);
      z = __builtin_amdgcn_wmma_f32_16x16x32_bf16(false, qa[1].v, false, kf[nt * 2 + 1].v,
                                                  (short)0, z, false, false);
      sacc[nt] = z;
    }

    // ---- causal mask (C/D layout: col = lane&15, row = r + 8*(lane>>4)) ----
#pragma unroll
    for (int nt = 0; nt < 2; ++nt) {
      int kvpos = kv0 + nt * 16 + rc;
#pragma unroll
      for (int r = 0; r < 8; ++r) {
        int s = s0 + r + (khalf << 3);
        sacc[nt][r] = (kvpos > CC + s) ? -3.0e38f : sacc[nt][r];
      }
    }

    // ---- online softmax: row stats via 16-lane xor reductions ----
    float alpha[8];
#pragma unroll
    for (int r = 0; r < 8; ++r) {
      float t = fmaxf(sacc[0][r], sacc[1][r]);
      for (int mm = 1; mm < 16; mm <<= 1) t = fmaxf(t, __shfl_xor(t, mm, 32));
      float nm = fmaxf(mrow[r], t);
      alpha[r] = __expf(mrow[r] - nm);
      mrow[r]  = nm;
      float p0 = __expf(sacc[0][r] - nm);
      float p1 = __expf(sacc[1][r] - nm);
      sacc[0][r] = p0; sacc[1][r] = p1;
      float rs = p0 + p1;
      for (int mm = 1; mm < 16; mm <<= 1) rs += __shfl_xor(rs, mm, 32);
      lrow[r] = lrow[r] * alpha[r] + rs;
    }

    // ---- relayout P (C-layout) -> A-fragment through LDS ----
#pragma unroll
    for (int r = 0; r < 8; ++r) {
      int prow = r + (khalf << 3);
      plds[prow * 32 + rc]      = f2bf(sacc[0][r]);
      plds[prow * 32 + 16 + rc] = f2bf(sacc[1][r]);
    }
    asm volatile("" ::: "memory");   // same-wave DS ops are in-order
    Frag pf;
    load_frag(pf, plds + rc * 32, khalf);  // LDS reads, pairs along kv contiguous

    // ---- wait for this tile's V DMA (8 next-tile DMAs may stay in flight) ----
    if (more) wait_async8(); else wait_async0();
#pragma unroll
    for (int dt = 0; dt < 4; ++dt) {
      Frag vf;
#pragma unroll
      for (int i = 0; i < 8; ++i) {
        int kb = ((i >> 2) << 4) + (khalf << 3) + ((i & 3) << 1);
        unsigned int lo = vcur[kb * 64 + dt * 16 + rc];
        unsigned int hi = vcur[(kb + 1) * 64 + dt * 16 + rc];
        vf.u[i] = lo | (hi << 16);
      }
      v8f c = oacc[dt];
#pragma unroll
      for (int r = 0; r < 8; ++r) c[r] *= alpha[r];
      c = __builtin_amdgcn_wmma_f32_16x16x32_bf16(false, pf.v, false, vf.v,
                                                  (short)0, c, false, false);
      oacc[dt] = c;
    }
  };

  // ---- pipelined kv loop (unrolled x2 so register buffers stay constant) ----
  Frag kfA[4], kfB[4];
  dmaV(0, vbase0);
  loadK(kfA, 0);
  for (int kv0 = 0; kv0 < kv_end; kv0 += 64) {
    bool m1 = (kv0 + 32 < kv_end);
    if (m1) loadK(kfB, kv0 + 32);
    process(kv0, kfA, vlds0, vbase1, m1);
    if (m1) {
      bool m2 = (kv0 + 64 < kv_end);
      if (m2) loadK(kfA, kv0 + 64);
      process(kv0 + 32, kfB, vlds1, vbase0, m2);
    }
  }

  // ---- finalize: divide by row sum, write bf16 ----
  unsigned short* obase = O + ((size_t)(b * SS + s0)) * EE + h * DD;
#pragma unroll
  for (int dt = 0; dt < 4; ++dt)
#pragma unroll
    for (int r = 0; r < 8; ++r) {
      int prow = r + (khalf << 3);
      obase[(size_t)prow * EE + dt * 16 + rc] = f2bf(oacc[dt][r] / lrow[r]);
    }
}

// ---------------- host launcher ----------------
extern "C" void kernel_launch(void* const* d_in, const int* in_sizes, int n_in,
                              void* d_out, int out_size, void* d_ws, size_t ws_size,
                              hipStream_t stream) {
  const float* x  = (const float*)d_in[0];
  const float* ck = (const float*)d_in[1];
  const float* cv = (const float*)d_in[2];
  const float* Wq = (const float*)d_in[3];
  const float* bq = (const float*)d_in[4];
  const float* Wk = (const float*)d_in[5];
  const float* bk = (const float*)d_in[6];
  const float* Wv = (const float*)d_in[7];
  const float* bv = (const float*)d_in[8];
  const float* Wo = (const float*)d_in[9];
  const float* bo = (const float*)d_in[10];

  const size_t nX  = (size_t)BB * SS * EE;   // 2,097,152
  const size_t nW  = (size_t)EE * EE;        //   262,144
  const size_t nKV = (size_t)BB * KVL * EE;  // 8,388,608

  unsigned short* ws  = (unsigned short*)d_ws;
  unsigned short* xbf = ws;            ws += nX;
  unsigned short* wbf = ws;            ws += 4 * nW;
  unsigned short* qbf = ws;            ws += nX;
  unsigned short* kbf = ws;            ws += nKV;
  unsigned short* vbf = ws;            ws += nKV;
  unsigned short* obf = ws;            ws += nX;

  dim3 blk(256);
  cvt_f32_bf16<<<1024, blk, 0, stream>>>(x, xbf, (long long)nX);
  cvt_f32_bf16<<<256, blk, 0, stream>>>(Wq, wbf + 0 * nW, (long long)nW);
  cvt_f32_bf16<<<256, blk, 0, stream>>>(Wk, wbf + 1 * nW, (long long)nW);
  cvt_f32_bf16<<<256, blk, 0, stream>>>(Wv, wbf + 2 * nW, (long long)nW);
  cvt_f32_bf16<<<256, blk, 0, stream>>>(Wo, wbf + 3 * nW, (long long)nW);
  cvt_cache<<<2048, blk, 0, stream>>>(ck, kbf, (long long)((size_t)BB * CC * EE));
  cvt_cache<<<2048, blk, 0, stream>>>(cv, vbf, (long long)((size_t)BB * CC * EE));

  const int M = BB * SS, N = EE, Kd = EE;
  const int tiles  = (M / 16) * (N / 64);   // 2048 wave-tiles (16x64 each)
  const int blocks = tiles / 8;             // 8 waves per block -> 256 blocks

  // q = (x Wq^T + bq) * D^-0.5  (fold attention scaling into Q)
  gemm_bf16<<<blocks, blk, 0, stream>>>(xbf, wbf + 0 * nW, bq, qbf, M, N, Kd, 1, 0.125f);
  // k_new / v_new scattered into rows [C, C+S) of the (B,KV,E) buffers
  gemm_bf16<<<blocks, blk, 0, stream>>>(xbf, wbf + 1 * nW, bk, kbf, M, N, Kd, 2, 1.0f);
  gemm_bf16<<<blocks, blk, 0, stream>>>(xbf, wbf + 2 * nW, bv, vbf, M, N, Kd, 2, 1.0f);

  // attention: B*H*(S/16) = 2048 waves = 256 blocks of 8 waves
  attn_kernel<<<256, blk, 0, stream>>>(qbf, kbf, vbf, obf);

  // final projection to f32 output
  gemm_bf16<<<blocks, blk, 0, stream>>>(obf, wbf + 3 * nW, bo, d_out, M, N, Kd, 0, 1.0f);
}
